// multichannel_GCN_44702019617103
// MI455X (gfx1250) — matrix operand order repaired and checked by reading, not compiled, not run on hardware
//
#include <hip/hip_runtime.h>
#include <hip/hip_bf16.h>
#include <stdint.h>

// ---- problem constants --------------------------------------------------
constexpr int BB   = 4;        // batch
constexpr int LL   = 256;      // seq len
constexpr int EE   = 200;      // embed dim
constexpr int EP   = 224;      // embed dim padded to 7*32 for bf16 WMMA K
constexpr int LP   = 256;      // padded l dim (16 tiles of 16)
constexpr int KB   = 7;        // EP/32 K-blocks
constexpr int NL   = 28415;    // labels
constexpr int NLP  = 28416;    // labels padded to 16
constexpr int NT   = 1776;     // NLP/16 label tiles
constexpr int NSTR = 5;        // streams
constexpr int SB   = 20;       // NSTR*BB
constexpr int COR  = 1000;
constexpr int LA   = 254;      // after conv1 (dil 1)
constexpr int LB   = 250;      // after conv2 (dil 2)
constexpr int LC   = 244;      // after conv3 (dil 3)
constexpr long NEDGE = 454640;

constexpr int TILE_U32 = 16 * EP / 2;   // 1792 u32 per staged l-tile (7168 B)

typedef __attribute__((ext_vector_type(16))) __bf16 v16bf;
typedef __attribute__((ext_vector_type(8)))  float  v8f;
typedef unsigned int v4u __attribute__((ext_vector_type(4)));

union Frag { unsigned int u[8]; v16bf v; };

// ---- helpers ------------------------------------------------------------
__device__ __forceinline__ unsigned short f2bf(float x) {
  unsigned int u = __float_as_uint(x);
  unsigned int r = u + 0x7FFFu + ((u >> 16) & 1u);
  return (unsigned short)(r >> 16);
}
__device__ __forceinline__ float seluf(float x) {
  const float scale = 1.0507009873554804934193349852946f;
  const float alpha = 1.6732632423543772848170429916717f;
  return x > 0.f ? scale * x : scale * alpha * (__expf(x) - 1.f);
}

// ---- generic zero (u32 granularity) ------------------------------------
__global__ __launch_bounds__(256) void k_zero(unsigned int* p, long n) {
  long i = (long)blockIdx.x * 256 + threadIdx.x;
  if (i < n) p[i] = 0u;
}

// ---- GCN edge aggregation: out[dst] += feat[src] ------------------------
__global__ __launch_bounds__(256) void k_edge_agg(const float* __restrict__ feat,
                                                  const int* __restrict__ src,
                                                  const int* __restrict__ dst,
                                                  float* __restrict__ out) {
  long i = (long)blockIdx.x * 256 + threadIdx.x;
  if (i >= NEDGE * (long)EE) return;
  int  e    = (int)(i % EE);
  long edge = i / EE;
  atomicAdd(out + (long)dst[edge] * EE + e, feat[(long)src[edge] * EE + e]);
}

// ---- GCN linear layer 1: relu(h @ w^T + b) ------------------------------
__global__ __launch_bounds__(256) void k_gcn_lin_relu(const float* __restrict__ hin,
                                                      const float* __restrict__ w,
                                                      const float* __restrict__ bias,
                                                      float* __restrict__ out) {
  __shared__ float row[EE];
  int n = blockIdx.x;
  for (int i = threadIdx.x; i < EE; i += 256) row[i] = hin[(long)n * EE + i];
  __syncthreads();
  int o = threadIdx.x;
  if (o < EE) {
    float acc = bias[o];
    const float* wr = w + (long)o * EE;
    for (int i = 0; i < EE; ++i) acc = fmaf(row[i], wr[i], acc);
    out[(long)n * EE + o] = fmaxf(acc, 0.f);
  }
}

// ---- GCN linear layer 2 + residual, emit bf16 padded rows ---------------
__global__ __launch_bounds__(256) void k_gcn_lin_resid_bf16(const float* __restrict__ hin,
                                                            const float* __restrict__ w,
                                                            const float* __restrict__ bias,
                                                            const float* __restrict__ resid,
                                                            unsigned short* __restrict__ out) {
  __shared__ float row[EE];
  int n = blockIdx.x;
  for (int i = threadIdx.x; i < EE; i += 256) row[i] = hin[(long)n * EE + i];
  __syncthreads();
  int o = threadIdx.x;
  if (o < EE) {
    float acc = bias[o];
    const float* wr = w + (long)o * EE;
    for (int i = 0; i < EE; ++i) acc = fmaf(row[i], wr[i], acc);
    acc += resid[(long)n * EE + o];
    out[(long)n * EP + o] = f2bf(acc);
  }
}

// ---- embedding lookup, [sb][l][e] layout --------------------------------
__global__ __launch_bounds__(256) void k_embed(const int* __restrict__ tok,
                                               const float* __restrict__ emb,
                                               float* __restrict__ x0) {
  int b = blockIdx.x / LL;
  int l = blockIdx.x % LL;
  int e = threadIdx.x;
  if (e < EE) {
    int t = tok[b * LL + l];
    x0[((long)b * LL + l) * EE + e] = emb[(long)t * EE + e];
  }
}

// ---- dilated conv1d (K=3) + SELU, f32 out -------------------------------
__global__ __launch_bounds__(256) void k_conv(const float* __restrict__ xin, int Lin, int Lout,
                                              int dil, const float* __restrict__ w,
                                              const float* __restrict__ bias,
                                              float* __restrict__ xout) {
  __shared__ float xs[3 * EE];
  int sb = blockIdx.x / Lout;
  int l  = blockIdx.x % Lout;
  for (int i = threadIdx.x; i < 3 * EE; i += 256) {
    int k = i / EE, c = i % EE;
    xs[i] = xin[((long)sb * Lin + l + k * dil) * EE + c];
  }
  __syncthreads();
  int o = threadIdx.x;
  if (o < EE) {
    float acc = bias[o];
    const float* wr = w + (long)o * EE * 3;
    #pragma unroll 4
    for (int i = 0; i < EE; ++i) {
      acc = fmaf(wr[i * 3 + 0], xs[i], acc);
      acc = fmaf(wr[i * 3 + 1], xs[EE + i], acc);
      acc = fmaf(wr[i * 3 + 2], xs[2 * EE + i], acc);
    }
    xout[((long)sb * Lout + l) * EE + o] = seluf(acc);
  }
}

// ---- conv3 (dil 3) + SELU, emit bf16 into zero-padded [sb][LP][EP] ------
__global__ __launch_bounds__(256) void k_conv3_bf16(const float* __restrict__ xin,
                                                    const float* __restrict__ w,
                                                    const float* __restrict__ bias,
                                                    unsigned short* __restrict__ x16) {
  __shared__ float xs[3 * EE];
  int sb = blockIdx.x / LC;
  int l  = blockIdx.x % LC;
  for (int i = threadIdx.x; i < 3 * EE; i += 256) {
    int k = i / EE, c = i % EE;
    xs[i] = xin[((long)sb * LB + l + k * 3) * EE + c];
  }
  __syncthreads();
  int o = threadIdx.x;
  if (o < EE) {
    float acc = bias[o];
    const float* wr = w + (long)o * EE * 3;
    #pragma unroll 4
    for (int i = 0; i < EE; ++i) {
      acc = fmaf(wr[i * 3 + 0], xs[i], acc);
      acc = fmaf(wr[i * 3 + 1], xs[EE + i], acc);
      acc = fmaf(wr[i * 3 + 2], xs[2 * EE + i], acc);
    }
    x16[((long)sb * LP + l) * EP + o] = f2bf(seluf(acc));
  }
}

// ---- fused attention: x_feat[b,n] = sum_s sum_l softmax_l(S)*S ----------
// Block = 8 waves, one batch element. Each wave owns TWO 16-label tiles
// (resident B fragments). All waves share identical A data, so the x l-tile
// (16 rows x 224 bf16 = 7 KB) is double-buffer staged in LDS cooperatively:
// coalesced global loads overlap WMMA compute; A fragments come from LDS.
// A = x tile (rows=l, K=e), B = lf^T (K=e, cols=n).
// D tile: row l = lt*16 + r + 8*half, col n = n0 + (lane&15).
__global__ __launch_bounds__(256) void k_attn(const unsigned short* __restrict__ lf16,
                                              const unsigned short* __restrict__ x16,
                                              float* __restrict__ xfeat) {
  __shared__ unsigned int sx32[2 * TILE_U32];   // 2 x 7168 B staging

  constexpr int NTB = NT / 16;                  // 111 tile-groups per batch
  int tid  = threadIdx.x;
  int wave = tid >> 5;
  int lane = tid & 31;
  int b    = blockIdx.x / NTB;
  int g    = blockIdx.x % NTB;
  int nt0  = (g * 8 + wave) * 2;                // even tile index
  int col  = lane & 15;
  int half = lane >> 4;

  // resident B fragments for both label tiles
  Frag Bf0[KB], Bf1[KB];
  const unsigned short* lfp0 = lf16 + (size_t)(nt0 * 16 + col) * EP + half * 16;
  const unsigned short* lfp1 = lfp0 + (size_t)16 * EP;
  #pragma unroll
  for (int kb = 0; kb < KB; ++kb) {
    *(v4u*)(Bf0[kb].u)     = *(const v4u*)(lfp0 + kb * 32);
    *(v4u*)(Bf0[kb].u + 4) = *(const v4u*)(lfp0 + kb * 32 + 8);
    *(v4u*)(Bf1[kb].u)     = *(const v4u*)(lfp1 + kb * 32);
    *(v4u*)(Bf1[kb].u + 4) = *(const v4u*)(lfp1 + kb * 32 + 8);
  }

  float xacc0 = 0.f, xacc1 = 0.f;
  for (int s = 0; s < NSTR; ++s) {
    const unsigned int* xsb = (const unsigned int*)(x16 + (size_t)(s * BB + b) * LP * EP);

    // preload tile 0 into buffer 0 (coalesced: 7 u32 per thread)
    {
      unsigned int r[7];
      #pragma unroll
      for (int i = 0; i < 7; ++i) r[i] = xsb[i * 256 + tid];
      #pragma unroll
      for (int i = 0; i < 7; ++i) sx32[i * 256 + tid] = r[i];
    }
    __syncthreads();

    float m0 = -3.0e38f, se0 = 0.f, swe0 = 0.f;
    float m1 = -3.0e38f, se1 = 0.f, swe1 = 0.f;

    for (int lt = 0; lt < 16; ++lt) {
      int cur = lt & 1;
      // issue global loads of next tile early (overlap with WMMA compute)
      unsigned int r[7];
      if (lt < 15) {
        const unsigned int* gnx = xsb + (size_t)(lt + 1) * TILE_U32;
        #pragma unroll
        for (int i = 0; i < 7; ++i) r[i] = gnx[i * 256 + tid];
      }

      // compute from staged tile: A fragments via LDS
      v8f c0 = {0.f, 0.f, 0.f, 0.f, 0.f, 0.f, 0.f, 0.f};
      v8f c1 = {0.f, 0.f, 0.f, 0.f, 0.f, 0.f, 0.f, 0.f};
      const unsigned short* sbase =
          (const unsigned short*)sx32 + (size_t)cur * (16 * EP) + col * EP + half * 16;
      #pragma unroll
      for (int kb = 0; kb < KB; ++kb) {
        Frag fa;
        *(v4u*)(fa.u)     = *(const v4u*)(sbase + kb * 32);
        *(v4u*)(fa.u + 4) = *(const v4u*)(sbase + kb * 32 + 8);
        c0 = __builtin_amdgcn_wmma_f32_16x16x32_bf16(
            false, fa.v, false, Bf0[kb].v, (short)0, c0, false, false);
        c1 = __builtin_amdgcn_wmma_f32_16x16x32_bf16(
            false, fa.v, false, Bf1[kb].v, (short)0, c1, false, false);
      }

      // tile-level softmax update: one rescale per tile, independent exps
      float sv0[8], sv1[8];
      #pragma unroll
      for (int rr = 0; rr < 8; ++rr) {
        int l = lt * 16 + rr + 8 * half;
        sv0[rr] = (l < LC) ? c0[rr] : -3.0e38f;
        sv1[rr] = (l < LC) ? c1[rr] : -3.0e38f;
      }
      float tm0 = sv0[0], tm1 = sv1[0];
      #pragma unroll
      for (int rr = 1; rr < 8; ++rr) {
        tm0 = fmaxf(tm0, sv0[rr]);
        tm1 = fmaxf(tm1, sv1[rr]);
      }
      float nm0 = fmaxf(m0, tm0), nm1 = fmaxf(m1, tm1);
      float sc0 = __expf(m0 - nm0), sc1 = __expf(m1 - nm1);
      se0 *= sc0; swe0 *= sc0;
      se1 *= sc1; swe1 *= sc1;
      #pragma unroll
      for (int rr = 0; rr < 8; ++rr) {
        float ev0 = __expf(sv0[rr] - nm0);
        float ev1 = __expf(sv1[rr] - nm1);
        se0 += ev0; swe0 = fmaf(ev0, sv0[rr], swe0);
        se1 += ev1; swe1 = fmaf(ev1, sv1[rr], swe1);
      }
      m0 = nm0; m1 = nm1;

      // stage next tile into the other buffer; one barrier per tile
      if (lt < 15) {
        unsigned int* ldst = sx32 + (size_t)(1 - cur) * TILE_U32;
        #pragma unroll
        for (int i = 0; i < 7; ++i) ldst[i * 256 + tid] = r[i];
      }
      __syncthreads();
    }

    // combine the two lane-halves holding the other 128 l positions
    {
      float m2   = __shfl_xor(m0, 16, 32);
      float se2  = __shfl_xor(se0, 16, 32);
      float swe2 = __shfl_xor(swe0, 16, 32);
      float nm = fmaxf(m0, m2);
      float t1 = __expf(m0 - nm), t2 = __expf(m2 - nm);
      xacc0 += (swe0 * t1 + swe2 * t2) / (se0 * t1 + se2 * t2);
    }
    {
      float m2   = __shfl_xor(m1, 16, 32);
      float se2  = __shfl_xor(se1, 16, 32);
      float swe2 = __shfl_xor(swe1, 16, 32);
      float nm = fmaxf(m1, m2);
      float t1 = __expf(m1 - nm), t2 = __expf(m2 - nm);
      xacc1 += (swe1 * t1 + swe2 * t2) / (se1 * t1 + se2 * t2);
    }
  }

  if (half == 0) {
    int n = nt0 * 16 + col;
    if (n < NL) xfeat[(size_t)b * NL + n] = xacc0;
    int n2 = (nt0 + 1) * 16 + col;
    if (n2 < NL) xfeat[(size_t)b * NL + n2] = xacc1;
  }
}

// ---- CorNet -------------------------------------------------------------
__global__ __launch_bounds__(256) void k_sigmoid(const float* __restrict__ x,
                                                 float* __restrict__ z, int n) {
  int i = blockIdx.x * 256 + threadIdx.x;
  if (i < n) z[i] = 1.f / (1.f + __expf(-x[i]));
}

__global__ __launch_bounds__(256) void k_cor_d(const float* __restrict__ z,
                                               const float* __restrict__ wd,
                                               const float* __restrict__ bd,
                                               float* __restrict__ c) {
  __shared__ float red[256];
  int b = blockIdx.x / COR;
  int o = blockIdx.x % COR;
  const float* zr = z + (size_t)b * NL;
  const float* wr = wd + (size_t)o * NL;
  float acc = 0.f;
  for (int n = threadIdx.x; n < NL; n += 256) acc = fmaf(zr[n], wr[n], acc);
  red[threadIdx.x] = acc;
  __syncthreads();
  for (int st = 128; st > 0; st >>= 1) {
    if (threadIdx.x < st) red[threadIdx.x] += red[threadIdx.x + st];
    __syncthreads();
  }
  if (threadIdx.x == 0) {
    float v = red[0] + bd[o];
    c[(size_t)b * COR + o] = v > 0.f ? v : (__expf(v) - 1.f);  // ELU
  }
}

__global__ __launch_bounds__(256) void k_cor_c(const float* __restrict__ c,
                                               const float* __restrict__ wc,
                                               const float* __restrict__ bc,
                                               const float* __restrict__ xin,
                                               float* __restrict__ xout) {
  __shared__ float cs[COR];
  constexpr int NBC = (NL + 255) / 256;
  int b = blockIdx.x / NBC;
  int n = (blockIdx.x % NBC) * 256 + threadIdx.x;
  for (int i = threadIdx.x; i < COR; i += 256) cs[i] = c[(size_t)b * COR + i];
  __syncthreads();
  if (n < NL) {
    float acc = bc[n] + xin[(size_t)b * NL + n];
    const float* wr = wc + (size_t)n * COR;
    #pragma unroll 4
    for (int o = 0; o < COR; ++o) acc = fmaf(cs[o], wr[o], acc);
    xout[(size_t)b * NL + n] = acc;
  }
}

// ---- launcher -----------------------------------------------------------
extern "C" void kernel_launch(void* const* d_in, const int* in_sizes, int n_in,
                              void* d_out, int out_size, void* d_ws, size_t ws_size,
                              hipStream_t stream) {
  const int* tok[5];
  for (int s = 0; s < 5; ++s) tok[s] = (const int*)d_in[s];
  const int*   src = (const int*)d_in[5];
  const int*   dst = (const int*)d_in[6];
  const float* gnf = (const float*)d_in[7];
  const float* emb = (const float*)d_in[8];
  const float* cw1 = (const float*)d_in[9];  const float* cb1 = (const float*)d_in[10];
  const float* cw2 = (const float*)d_in[11]; const float* cb2 = (const float*)d_in[12];
  const float* cw3 = (const float*)d_in[13]; const float* cb3 = (const float*)d_in[14];
  const float* gw1 = (const float*)d_in[15]; const float* gb1 = (const float*)d_in[16];
  const float* gw2 = (const float*)d_in[17]; const float* gb2 = (const float*)d_in[18];
  const float* wd1 = (const float*)d_in[19]; const float* bd1 = (const float*)d_in[20];
  const float* wc1 = (const float*)d_in[21]; const float* bc1 = (const float*)d_in[22];
  const float* wd2 = (const float*)d_in[23]; const float* bd2 = (const float*)d_in[24];
  const float* wc2 = (const float*)d_in[25]; const float* bc2 = (const float*)d_in[26];

  char* ws = (char*)d_ws;
  size_t off = 0;
  auto alloc = [&](size_t bytes) -> void* {
    void* p = ws + off;
    off = (off + bytes + 255) & ~(size_t)255;
    return p;
  };
  float*          hA   = (float*)alloc((size_t)NLP * EE * 4);
  float*          hB   = (float*)alloc((size_t)NLP * EE * 4);
  unsigned short* lf16 = (unsigned short*)alloc((size_t)NLP * EP * 2);
  float*          x0   = (float*)alloc((size_t)SB * LL * EE * 4);
  float*          x1   = (float*)alloc((size_t)SB * LA * EE * 4);
  float*          x2   = (float*)alloc((size_t)SB * LB * EE * 4);
  unsigned short* x16  = (unsigned short*)alloc((size_t)SB * LP * EP * 2);
  float*          xf   = (float*)alloc((size_t)BB * NL * 4);
  float*          zb   = (float*)alloc((size_t)BB * NL * 4);
  float*          cb   = (float*)alloc((size_t)BB * COR * 4);

  auto Z = [&](void* p, long n32) {
    k_zero<<<(unsigned)((n32 + 255) / 256), 256, 0, stream>>>((unsigned int*)p, n32);
  };

  // ---- LabelNet (GCN x2) ----
  long ethr = NEDGE * (long)EE;
  Z(hA, (long)NLP * EE);
  k_edge_agg<<<(unsigned)((ethr + 255) / 256), 256, 0, stream>>>(gnf, src, dst, hA);
  k_gcn_lin_relu<<<NL, 256, 0, stream>>>(hA, gw1, gb1, hB);
  Z(hA, (long)NLP * EE);
  k_edge_agg<<<(unsigned)((ethr + 255) / 256), 256, 0, stream>>>(hB, src, dst, hA);
  Z(lf16, (long)NLP * EP / 2);
  k_gcn_lin_resid_bf16<<<NL, 256, 0, stream>>>(hA, gw2, gb2, gnf, lf16);

  // ---- streams: embed + 3 dilated convs -> bf16 padded ----
  Z(x16, (long)SB * LP * EP / 2);
  for (int s = 0; s < NSTR; ++s)
    k_embed<<<BB * LL, 256, 0, stream>>>(tok[s], emb, x0 + (size_t)s * BB * LL * EE);
  k_conv<<<SB * LA, 256, 0, stream>>>(x0, LL, LA, 1, cw1, cb1, x1);
  k_conv<<<SB * LB, 256, 0, stream>>>(x1, LA, LB, 2, cw2, cb2, x2);
  k_conv3_bf16<<<SB * LC, 256, 0, stream>>>(x2, cw3, cb3, x16);

  // ---- fused attention (WMMA, LDS double-buffered A staging) ----
  k_attn<<<(NT / 16) * BB, 256, 0, stream>>>(lf16, x16, xf);

  // ---- CorNet x2 ----
  int nxf = BB * NL;
  constexpr int NBC = (NL + 255) / 256;
  k_sigmoid<<<(nxf + 255) / 256, 256, 0, stream>>>(xf, zb, nxf);
  k_cor_d<<<BB * COR, 256, 0, stream>>>(zb, wd1, bd1, cb);
  k_cor_c<<<BB * NBC, 256, 0, stream>>>(cb, wc1, bc1, xf, xf);
  k_sigmoid<<<(nxf + 255) / 256, 256, 0, stream>>>(xf, zb, nxf);
  k_cor_d<<<BB * COR, 256, 0, stream>>>(zb, wd2, bd2, cb);
  k_cor_c<<<BB * NBC, 256, 0, stream>>>(cb, wc2, bc2, xf, (float*)d_out);
}
